// MiniRocketFeatures_36807869726976
// MI455X (gfx1250) — matrix operand order repaired and checked by reading, not compiled, not run on hardware
//
#include <hip/hip_runtime.h>
#include <hip/hip_bf16.h>
#include <cmath>

#define SEQ   2048
#define NK    84      // kernels
#define CIN   9       // input channels
#define KW    9       // taps per kernel
#define BATCH 64
#define MAXD  32
#define MAXF  16      // max features per (dilation, kernel)
#define KPAD  96      // K = 81 padded to 3 chunks of 32
#define NPAD  96      // 84 kernels padded to 6 tiles of 16

typedef __attribute__((ext_vector_type(16))) _Float16 v16h;
typedef __attribute__((ext_vector_type(8)))  float    v8f;
typedef __attribute__((ext_vector_type(8)))  unsigned v8u;

struct Sched {
    int D;
    int total_feat;       // sum over i of NK * F_i  (= 9996)
    int max_f;            // bias tensor stride
    int halo;             // 4 * max dilation (floats)
    int stridew;          // SEQ + 2*halo (words per channel row in LDS)
    int dil[MAXD];
    int nfpd[MAXD];
    int pad[MAXD];
    int off[MAXD];        // output feature offset of dilation block i
};

// ---------------------------------------------------------------------------
// Prep: W_i[n][K] = cc[i,c,n] * kernel[n,j]  (values in {0,-1,2}: exact f16)
// Layout: wt[((i*NPAD)+n)*KPAD + K], K = c*KW + j, zero padded (K>=81 -> 0).
// ---------------------------------------------------------------------------
__global__ void build_w_kernel(const float* __restrict__ kernels,
                               const float* __restrict__ cc,
                               _Float16* __restrict__ wt, int D) {
    int idx = blockIdx.x * blockDim.x + threadIdx.x;
    int total = D * NPAD * KPAD;
    if (idx >= total) return;
    int i   = idx / (NPAD * KPAD);
    int rem = idx % (NPAD * KPAD);
    int n   = rem / KPAD;
    int K   = rem % KPAD;
    float v = 0.f;
    if (n < NK && K < CIN * KW) {
        int c = K / KW, j = K % KW;
        float w = kernels[((size_t)(c * NK + n)) * KW + j]; // tiled (756,1,9)
        float m = cc[((size_t)i * CIN + c) * NK + n];       // (D,9,84,1)
        v = w * m;
    }
    wt[idx] = (_Float16)v;
}

// ---------------------------------------------------------------------------
// Main: per (dilation i, batch b) workgroup; 6 waves, wave w owns 16 kernels.
// x staged in LDS as packed {hi,lo} f16 words with a zeroed halo => gathers
// are plain ds_load_b32; A fragments assembled with v_perm_b32.
// ---------------------------------------------------------------------------
__global__ void __launch_bounds__(192)
minirocket_kernel(const float* __restrict__ x,
                  const _Float16* __restrict__ wt,
                  const float* __restrict__ biases,
                  float* __restrict__ out,
                  Sched sc) {
    extern __shared__ unsigned smem[];
    const int i    = blockIdx.x;           // dilation index
    const int b    = blockIdx.y;           // batch row
    const int tid  = threadIdx.x;
    const int wave = tid >> 5;             // kernel tile 0..5
    const int lane = tid & 31;
    const int d    = sc.dil[i];
    const int p    = sc.pad[i];
    const int Fi   = sc.nfpd[i];
    const int p1   = i & 1;
    const int HF   = sc.halo;
    const int SW   = sc.stridew;

    unsigned* sbias = smem;                         // NK*max_f raw f32 words
    unsigned* sxp   = smem + NK * sc.max_f;         // CIN*SW packed {hi,lo}

    // --- async-stage this dilation's bias rows into LDS (ASYNCcnt path) ----
    {
        const float* bsrc = biases + (size_t)i * NK * sc.max_f;
        const int nchunk = (NK * sc.max_f) >> 2;    // 16B chunks, 84*max_f%4==0
        for (int q = tid; q < nchunk; q += 192) {
            unsigned ldsAddr = (unsigned)(size_t)(void*)(sbias + q * 4);
            unsigned gOff    = (unsigned)(q * 16);
            asm volatile("global_load_async_to_lds_b128 %0, %1, %2"
                         :: "v"(ldsAddr), "v"(gOff), "s"(bsrc) : "memory");
        }
    }

    // --- stage x(b): convert f32 -> packed (hi:f16 | lo:f16) words ---------
    {
        const float4* g4 = (const float4*)(x + (size_t)b * CIN * SEQ);
        for (int q = tid; q < CIN * SEQ / 4; q += 192) {
            float4 v4 = g4[q];
            int c = q / (SEQ / 4), col = (q % (SEQ / 4)) * 4;
            unsigned* dst = sxp + c * SW + HF + col;   // 16B aligned
            float vv[4] = {v4.x, v4.y, v4.z, v4.w};
            uint4 w;
            unsigned wo[4];
            #pragma unroll
            for (int e = 0; e < 4; ++e) {
                _Float16 h = (_Float16)vv[e];
                _Float16 l = (_Float16)(vv[e] - (float)h);
                wo[e] = (unsigned)__builtin_bit_cast(unsigned short, h)
                      | ((unsigned)__builtin_bit_cast(unsigned short, l) << 16);
            }
            w.x = wo[0]; w.y = wo[1]; w.z = wo[2]; w.w = wo[3];
            *(uint4*)dst = w;
        }
        // zero only the +/-4d halo this block will actually read
        const int zl = 4 * d;
        for (int q = tid; q < CIN * zl; q += 192) {
            int c = q / zl, o = q % zl;
            sxp[c * SW + HF - zl + o] = 0u;
            sxp[c * SW + HF + SEQ + o] = 0u;
        }
    }
    asm volatile("s_wait_asynccnt 0" ::: "memory");
    __syncthreads();

    const int nloc  = lane & 15;           // column within 16-kernel tile
    const int hi16  = lane >> 4;           // lane half (A/B K-split)
    const int k     = wave * 16 + nloc;    // kernel id (0..95; >=84 invalid)
    const bool kval = (k < NK);
    const bool full = ((k & 1) == p1);     // full-range vs cropped parity

    // B fragments (3 K-chunks) in VGPRs for the whole wave lifetime.
    // Dense 16-bit B layout: lanes0-15: N=lane, K=0..7,16..23; lanes16-31: K+8.
    v16h Bfrag[3];
    {
        const _Float16* wrow = wt + ((size_t)i * NPAD + k) * KPAD;
        #pragma unroll
        for (int ch = 0; ch < 3; ++ch)
            #pragma unroll
            for (int e = 0; e < 16; ++e)
                Bfrag[ch][e] = wrow[ch * 32 + hi16 * 8 + ((e < 8) ? e : e + 8)];
    }

    // Per-feature biases in registers (+inf pads never count)
    float bias_r[MAXF];
    #pragma unroll
    for (int f = 0; f < MAXF; ++f)
        bias_r[f] = (kval && f < Fi)
                  ? __builtin_bit_cast(float, sbias[k * sc.max_f + f])
                  : __builtin_bit_cast(float, 0x7f800000u);

    int cnt[MAXF];
    #pragma unroll
    for (int f = 0; f < MAXF; ++f) cnt[f] = 0;

    const int m = lane & 15;               // A-matrix row (time within tile)
    const float NEG_INF = __builtin_bit_cast(float, 0xff800000u);

    for (int tile = 0; tile < SEQ / 16; ++tile) {
        const int tm = tile * 16 + m;      // global time of this A row
        v8f acc = {};
        #pragma unroll
        for (int ch = 0; ch < 3; ++ch) {
            v8u ahi, alo;
            #pragma unroll
            for (int v = 0; v < 8; ++v) {
                int kk = ch * 32 + ((v < 4) ? 2 * v : 16 + 2 * (v - 4)) + hi16 * 8;
                // K>=81 rows of W are zero, so clamp K (chunk 2 only): 0*x==0
                int K0 = (ch == 2 && kk     > 80) ? 80 : kk;
                int K1 = (ch == 2 && kk + 1 > 80) ? 80 : kk + 1;
                int c0 = K0 / KW, j0 = K0 % KW;
                int c1 = K1 / KW, j1 = K1 % KW;
                unsigned w0 = sxp[c0 * SW + HF + (j0 - 4) * d + tm];
                unsigned w1 = sxp[c1 * SW + HF + (j1 - 4) * d + tm];
                ahi[v] = __builtin_amdgcn_perm(w1, w0, 0x05040100u); // hi pair
                alo[v] = __builtin_amdgcn_perm(w1, w0, 0x07060302u); // lo pair
            }
            acc = __builtin_amdgcn_wmma_f32_16x16x32_f16(
                      false, __builtin_bit_cast(v16h, ahi), false, Bfrag[ch],
                      (short)0, acc, false, false);
            acc = __builtin_amdgcn_wmma_f32_16x16x32_f16(
                      false, __builtin_bit_cast(v16h, alo), false, Bfrag[ch],
                      (short)0, acc, false, false);
        }
        // PPV: C/D layout VGPR r: M = r + 8*hi16, N = lane&15. Branchless crop.
        #pragma unroll
        for (int r = 0; r < 8; ++r) {
            int t = (tm - m) + r + hi16 * 8;
            bool tv = full || (t >= p && t < SEQ - p);
            float cv = tv ? acc[r] : NEG_INF;
            #pragma unroll
            for (int f = 0; f < MAXF; ++f)
                if (f < Fi) cnt[f] += (cv > bias_r[f]) ? 1 : 0;
        }
    }

    // Reduce the two lane-halves (same kernel column) and emit features.
    const float invLen = full ? (1.0f / SEQ) : (1.0f / (SEQ - 2 * p));
    #pragma unroll
    for (int f = 0; f < MAXF; ++f) {
        int tot = cnt[f] + __shfl_xor(cnt[f], 16, 32);
        if (hi16 == 0 && kval && f < Fi) {
            int pos = sc.off[i] + (full ? 0 : (NK / 2) * Fi) + (k >> 1) * Fi + f;
            out[(size_t)b * sc.total_feat + pos] = (float)tot * invLen;
        }
    }
}

// ---------------------------------------------------------------------------
extern "C" void kernel_launch(void* const* d_in, const int* in_sizes, int n_in,
                              void* d_out, int out_size, void* d_ws, size_t ws_size,
                              hipStream_t stream) {
    const float* x       = (const float*)d_in[0];
    const float* kernels = (const float*)d_in[1];
    const float* cc      = (const float*)d_in[2];
    const float* biases  = (const float*)d_in[3];
    float* out = (float*)d_out;

    const int D_in  = in_sizes[2] / (CIN * NK);
    const int max_f = in_sizes[3] / (NK * (D_in > 0 ? D_in : 1));

    // Recompute the reference schedule deterministically on the host.
    Sched sc;
    {
        const double max_exp = log2(2047.0 / 8.0);     // log2((SEQ-1)/(KW-1))
        int vals[32];
        for (int t = 0; t < 32; ++t)
            vals[t] = (int)exp2(max_exp * (double)t / 31.0);  // trunc == astype
        int D = 0, counts[MAXD];
        for (int t = 0; t < 32; ++t) {
            if (D > 0 && vals[t] == sc.dil[D - 1]) counts[D - 1]++;
            else { sc.dil[D] = vals[t]; counts[D] = 1; D++; }
        }
        const double mult = 119.0 / 32.0;   // (10000//84) / true_max
        int sum = 0;
        for (int t = 0; t < D; ++t) { sc.nfpd[t] = (int)(counts[t] * mult); sum += sc.nfpd[t]; }
        int rem = 119 - sum, ri = 0;
        while (rem > 0) { sc.nfpd[ri]++; rem--; ri = (ri + 1) % D; }
        int off = 0;
        for (int t = 0; t < D; ++t) {
            sc.pad[t] = (KW - 1) * sc.dil[t] / 2;
            sc.off[t] = off;
            off += NK * sc.nfpd[t];
        }
        if (D > D_in && D_in > 0) D = D_in;   // never index inputs OOB
        sc.D = D; sc.total_feat = off; sc.max_f = max_f;
        sc.halo = 4 * sc.dil[D - 1];          // dilations sorted ascending
        sc.stridew = SEQ + 2 * sc.halo;
    }

    // Build exact-f16 combined weight matrices in workspace.
    _Float16* wt = (_Float16*)d_ws;
    const int totalW = sc.D * NPAD * KPAD;
    hipLaunchKernelGGL(build_w_kernel, dim3((totalW + 255) / 256), dim3(256), 0,
                       stream, kernels, cc, wt, sc.D);

    const size_t shmem = (size_t)(NK * sc.max_f + CIN * sc.stridew) * 4u;
    dim3 grid(sc.D, BATCH);
    hipLaunchKernelGGL(minirocket_kernel, grid, dim3(192), shmem, stream,
                       x, wt, biases, out, sc);
}
